// GPSLayer_53077205844584
// MI455X (gfx1250) — matrix operand order repaired and checked by reading, not compiled, not run on hardware
//
#include <hip/hip_runtime.h>
#include <hip/hip_bf16.h>
#include <math.h>
#include <stdint.h>

typedef __bf16 bf16;
typedef __attribute__((ext_vector_type(16))) __bf16 v16bf;
typedef __attribute__((ext_vector_type(8)))  float  v8f;
typedef unsigned int u32x4 __attribute__((ext_vector_type(4)));
typedef int i32x8 __attribute__((ext_vector_type(8)));
typedef int i32x4 __attribute__((ext_vector_type(4)));

#define D_DIM 256
#define H_DIM 8
#define B_DIM 64
#define N_DIM 512
#define NN_DIM 32768
#define E_DIM 524288

#if defined(__has_builtin)
#if __has_builtin(__builtin_amdgcn_tensor_load_to_lds) && \
    __has_builtin(__builtin_amdgcn_s_wait_tensorcnt)
#define USE_TDM 1
#endif
#endif

// Dynamic LDS (offset 0 => TDM lds_addr is a plain byte offset)
extern __shared__ char dynsmem[];

// ---------------------------------------------------------------- WMMA helpers
__device__ inline v8f wmma_bf16(v16bf a, v16bf b, v8f c) {
  return __builtin_amdgcn_wmma_f32_16x16x32_bf16(false, a, false, b, (short)0, c,
                                                 false, false);
}

// A-matrix fragment (16x32 bf16, ISA 7.12.2). Row pitch must be 16B aligned.
// lane m = lane&15; khalf = 8*(lane>=16): bytes [khalf*2, +16) and [32+khalf*2, +16)
__device__ inline v16bf load_fragA(const bf16* base, int row, int stride) {
  int lane = threadIdx.x & 31;
  const char* r = (const char*)(base + (size_t)(row + (lane & 15)) * stride);
  int k0b = (lane >> 4) << 4;  // 0 or 16 bytes
  union { v16bf v; uint4 q[2]; } f;
  f.q[0] = *(const uint4*)(r + k0b);
  f.q[1] = *(const uint4*)(r + 32 + k0b);
  return f.v;
}

// B-matrix fragment (32x16 bf16), source stored N-major (row=n, cols=k):
// n = lane&15; khalf = 8*(lane>=16): bytes [khalf*2, +16) and [32+khalf*2, +16)
__device__ inline v16bf load_fragB(const bf16* base, int row, int stride) {
  int lane = threadIdx.x & 31;
  const char* r = (const char*)(base + (size_t)(row + (lane & 15)) * stride);
  int k0b = (lane >> 4) << 4;  // 0 or 16 bytes
  union { v16bf v; uint4 q[2]; } f;
  f.q[0] = *(const uint4*)(r + k0b);
  f.q[1] = *(const uint4*)(r + 32 + k0b);
  return f.v;
}

#ifdef USE_TDM
// TDM: load a rows x rowElems (bf16) tile, global row stride = strideElems, into
// LDS at byte offset ldsOff with +4dw padding every 16dw => LDS pitch 40 bf16.
__device__ inline void tdm_load_2d(const void* gaddr, unsigned ldsOff,
                                   int rowElems, int rows, int strideElems) {
  unsigned long long ga = (unsigned long long)(uintptr_t)gaddr;
  u32x4 g0;
  g0.x = 1u;                                   // count=1, user descriptor
  g0.y = ldsOff;                               // lds_addr (bytes)
  g0.z = (unsigned)ga;                         // global_addr[31:0]
  g0.w = (unsigned)((ga >> 32) & 0x01FFFFFFu) | (2u << 30);  // addr[56:32]|type=2
  i32x8 g1;
  g1[0] = (1 << 16) | (1 << 20) | (3 << 22) | (3 << 25);
  //        data=2B    pad_en     16dw ival     4dw amount
  g1[1] = (rowElems & 0xFFFF) << 16;                          // tensor_dim0 lo
  g1[2] = ((rows & 0xFFFF) << 16) | ((rowElems >> 16) & 0xFFFF);  // dim0 hi|dim1 lo
  g1[3] = ((rowElems & 0xFFFF) << 16) | ((rows >> 16) & 0xFFFF);  // dim1 hi|tile0
  g1[4] = rows & 0xFFFF;                                      // tile_dim1, tile2=0
  g1[5] = strideElems;                                        // dim0_stride lo32
  g1[6] = 0;
  g1[7] = 0;
  i32x4 z4 = {0, 0, 0, 0};
  i32x8 z8 = {0, 0, 0, 0, 0, 0, 0, 0};
  // 6-arg form (this toolchain): (g0, g1, g2, g3, g4, cpol)
  __builtin_amdgcn_tensor_load_to_lds(g0, g1, z4, z4, z8, 0);
}
#endif

// ---------------------------------------------------------------- small kernels
__global__ void zero_f(float* p, size_t n) {
  size_t i = (size_t)blockIdx.x * 256 + threadIdx.x;
  if (i < n) p[i] = 0.f;
}

__global__ void f2b_kernel(const float* s, bf16* d, int n) {
  int i = blockIdx.x * 256 + threadIdx.x;
  if (i < n) d[i] = (bf16)s[i];
}

__global__ void edge_scalar_kernel(const int* ei, const float* ew, float* sumew,
                                   float* cnt) {
  int e = blockIdx.x * 256 + threadIdx.x;
  if (e < E_DIM) {
    int d = ei[E_DIM + e];
    atomicAdd(&sumew[d], ew[e]);
    atomicAdd(&cnt[d], 1.0f);
  }
}

__global__ void edge_scatter_kernel(const int* ei, const float* X, float* agg) {
  int e = blockIdx.x;
  int c = threadIdx.x;
  int s = ei[e], d = ei[E_DIM + e];
  atomicAdd(&agg[(size_t)d * D_DIM + c], X[(size_t)s * D_DIM + c]);
}

__global__ void make_h0_kernel(const float* agg, const float* sumew,
                               const float* cnt, const float* eww,
                               const float* ewb, const float* X, bf16* H0) {
  int n = blockIdx.x, c = threadIdx.x;
  size_t idx = (size_t)n * D_DIM + c;
  float v = agg[idx] + sumew[n] * eww[c] + cnt[n] * ewb[c] + X[idx];
  H0[idx] = (bf16)v;
}

__global__ void pe_add_kernel(const float* X, bf16* Y) {
  int row = blockIdx.x, col = threadIdx.x;
  size_t idx = (size_t)row * D_DIM + col;
  int pos = row % N_DIM;
  int i2 = col & ~1;
  float div = __expf((float)i2 * (-9.210340372f / (float)D_DIM));
  float ang = (float)pos * div;
  float pe = (col & 1) ? __cosf(ang) : __sinf(ang);
  Y[idx] = (bf16)(X[idx] + pe);
}

__global__ void repack_qkv_kernel(const float* qkv, bf16* Qb, bf16* Kb, bf16* Vb) {
  int row = blockIdx.x, col = threadIdx.x;
  int b = row / N_DIM, n = row % N_DIM;
  int h = col >> 5, d = col & 31;
  size_t src = (size_t)row * (3 * D_DIM);
  size_t dst = (((size_t)b * H_DIM + h) * N_DIM + n) * 32 + d;
  const float scale = 0.17677669529f;  // 1/sqrt(32)
  Qb[dst] = (bf16)(qkv[src + col] * scale);
  Kb[dst] = (bf16)(qkv[src + D_DIM + col]);
  Vb[dst] = (bf16)(qkv[src + 2 * D_DIM + col]);
}

__global__ __launch_bounds__(256) void residual_ln_kernel(
    const float* A, const float* Bt, const float* g, const float* be, float* Yf,
    bf16* Yb) {
  __shared__ float red[256];
  int row = blockIdx.x, c = threadIdx.x;
  size_t off = (size_t)row * D_DIM + c;
  float v = A[off] + Bt[off];
  red[c] = v;
  __syncthreads();
  for (int s = 128; s > 0; s >>= 1) {
    if (c < s) red[c] += red[c + s];
    __syncthreads();
  }
  float mean = red[0] * (1.f / D_DIM);
  __syncthreads();
  float d = v - mean;
  red[c] = d * d;
  __syncthreads();
  for (int s = 128; s > 0; s >>= 1) {
    if (c < s) red[c] += red[c + s];
    __syncthreads();
  }
  float var = red[0] * (1.f / D_DIM);
  float y = d * rsqrtf(var + 1e-5f) * g[c] + be[c];
  if (Yf) Yf[off] = y;
  if (Yb) Yb[off] = (bf16)y;
}

// ---------------------------------------------------------------- WMMA GEMM
// C(MxN) = act(A(MxK bf16) @ W(NxK bf16)^T + bias); 128x128 tile, 8 waves.
// Dynamic LDS: As @0 (128x40 bf16 = 10240B), Bs @10240. Total 20480B.
#define GEMM_LDS_BYTES 20480
__global__ __launch_bounds__(256) void gemm_bf16_kernel(
    const bf16* __restrict__ A, const bf16* __restrict__ W,
    const float* __restrict__ bias, float* Cf, bf16* Cb, int M, int N, int K,
    int act) {
  bf16* As = (bf16*)dynsmem;
  bf16* Bs = (bf16*)(dynsmem + 10240);
  int tileN = blockIdx.x * 128;
  int tileM = blockIdx.y * 128;
  int wave = threadIdx.x >> 5;
  int lane = threadIdx.x & 31;

  v8f acc[8];
#pragma unroll
  for (int i = 0; i < 8; ++i) acc[i] = (v8f){};

  for (int k0 = 0; k0 < K; k0 += 32) {
    __syncthreads();  // previous tiles fully consumed
#ifdef USE_TDM
    if (wave == 0) {
      tdm_load_2d(A + (size_t)tileM * K + k0, 0u, 32, 128, K);
      tdm_load_2d(W + (size_t)tileN * K + k0, 10240u, 32, 128, K);
      __builtin_amdgcn_s_wait_tensorcnt(0);
    }
#else
    {
      int t8 = threadIdx.x * 8;
      int row = t8 >> 5, col = t8 & 31;
#pragma unroll
      for (int p = 0; p < 2; ++p) {
        int rr = row + p * 64;
        uint4 da = *(const uint4*)(A + (size_t)(tileM + rr) * K + k0 + col);
        *(uint4*)(As + rr * 40 + col) = da;
        uint4 db = *(const uint4*)(W + (size_t)(tileN + rr) * K + k0 + col);
        *(uint4*)(Bs + rr * 40 + col) = db;
      }
    }
#endif
    __syncthreads();
    v16bf af = load_fragA(As, wave * 16, 40);
#pragma unroll
    for (int nt = 0; nt < 8; ++nt) {
      v16bf bf = load_fragB(Bs, nt * 16, 40);
      acc[nt] = wmma_bf16(af, bf, acc[nt]);
    }
  }

  int coln = lane & 15;
  int r0 = (lane >> 4) << 3;
#pragma unroll
  for (int nt = 0; nt < 8; ++nt) {
    int n = tileN + nt * 16 + coln;
    float bv = bias ? bias[n] : 0.f;
#pragma unroll
    for (int r = 0; r < 8; ++r) {
      float v = acc[nt][r] + bv;
      if (act == 1) v = fmaxf(v, 0.f);
      else if (act == 2) v = 0.5f * v * (1.f + erff(v * 0.70710678118f));
      size_t off = (size_t)(tileM + wave * 16 + r0 + r) * N + n;
      if (Cf) Cf[off] = v;
      if (Cb) Cb[off] = (bf16)v;
    }
  }
}

// ---------------------------------------------------------------- flash attention
// grid (B*H, N/128), 256 threads; wave w owns 16 query rows; dh=32 = one WMMA K.
// Dynamic LDS layout (bytes):
//   Ks    @0      64x40 bf16  = 5120
//   Vt    @5120   32x72 bf16  = 4608
//   Ssc   @9728   8 x 16x66 f32 = 33792
//   Pp    @43520  8 x 16x72 bf16 = 18432
//   alpha @61952  8x16 f32 = 512
//   sums  @62464  8x16 f32 = 512     total 62976
#define ATTN_LDS_BYTES 62976
__global__ __launch_bounds__(256) void attn_kernel(const bf16* __restrict__ Qb,
                                                   const bf16* __restrict__ Kb,
                                                   const bf16* __restrict__ Vb,
                                                   bf16* __restrict__ Out) {
  bf16* Ks = (bf16*)dynsmem;
  bf16* Vt = (bf16*)(dynsmem + 5120);
  float* Ssc = (float*)(dynsmem + 9728);
  bf16* Pp = (bf16*)(dynsmem + 43520);
  float* alphaS = (float*)(dynsmem + 61952);
  float* sumS = (float*)(dynsmem + 62464);

  int bh = blockIdx.x;
  int qBlk = blockIdx.y * 128;
  int wave = threadIdx.x >> 5;
  int lane = threadIdx.x & 31;
  int qRow0 = qBlk + wave * 16;

  const bf16* qbase = Qb + ((size_t)bh * N_DIM + qRow0) * 32;
  v16bf qfrag = load_fragA(qbase, 0, 32);

  v8f oacc[2];
  oacc[0] = (v8f){}; oacc[1] = (v8f){};
  float mrow = -1e30f, lrow = 0.f;  // live in lanes 0..15

  for (int kb = 0; kb < N_DIM; kb += 64) {
    // ---- stage K (TDM if available) and V^T (manual transpose) into LDS
#ifdef USE_TDM
    if (wave == 0)
      tdm_load_2d(Kb + ((size_t)bh * N_DIM + kb) * 32, 0u, 32, 64, 32);
#else
    {
      int t8 = threadIdx.x * 8;
      int row = t8 >> 5, colk = t8 & 31;
      uint4 dk = *(const uint4*)(Kb + ((size_t)bh * N_DIM + kb + row) * 32 + colk);
      *(uint4*)(Ks + row * 40 + colk) = dk;
    }
#endif
    {
      int n = threadIdx.x >> 2;
      int db = (threadIdx.x & 3) * 8;
      uint4 dv = *(const uint4*)(Vb + ((size_t)bh * N_DIM + kb + n) * 32 + db);
      const bf16* pv = (const bf16*)&dv;
#pragma unroll
      for (int i = 0; i < 8; ++i) Vt[(db + i) * 72 + n] = pv[i];
    }
#ifdef USE_TDM
    if (wave == 0) __builtin_amdgcn_s_wait_tensorcnt(0);
#endif
    __syncthreads();

    // ---- S = Q @ K^T for this wave's 16 rows x 64 keys
#pragma unroll
    for (int nt = 0; nt < 4; ++nt) {
      v16bf kf = load_fragB(Ks, nt * 16, 40);
      v8f s = wmma_bf16(qfrag, kf, (v8f){});
      int coln = nt * 16 + (lane & 15);
      int r0 = (lane >> 4) << 3;
#pragma unroll
      for (int r = 0; r < 8; ++r)
        Ssc[wave * 1056 + (r0 + r) * 66 + coln] = s[r];
    }
    __syncthreads();

    // ---- online softmax: lane l (<16) owns query row l of this wave
    if (lane < 16) {
      float* srow = &Ssc[wave * 1056 + lane * 66];
      float bm = -1e30f;
      for (int j = 0; j < 64; ++j) bm = fmaxf(bm, srow[j]);
      float mn = fmaxf(mrow, bm);
      float alpha = __expf(mrow - mn);
      float ps = 0.f;
      bf16* prow = &Pp[wave * 1152 + lane * 72];
      for (int j = 0; j < 64; ++j) {
        float p = __expf(srow[j] - mn);
        ps += p;
        prow[j] = (bf16)p;
      }
      lrow = lrow * alpha + ps;
      mrow = mn;
      alphaS[wave * 16 + lane] = alpha;
    }
    __syncthreads();

    // ---- rescale O and accumulate P @ V
    {
      int r0 = (lane >> 4) << 3;
      float al[8];
#pragma unroll
      for (int r = 0; r < 8; ++r) al[r] = alphaS[wave * 16 + r0 + r];
#pragma unroll
      for (int dt = 0; dt < 2; ++dt)
#pragma unroll
        for (int r = 0; r < 8; ++r) oacc[dt][r] *= al[r];
#pragma unroll
      for (int ks = 0; ks < 2; ++ks) {
        v16bf pf = load_fragA(&Pp[wave * 1152 + ks * 32], 0, 72);
#pragma unroll
        for (int dt = 0; dt < 2; ++dt) {
          v16bf vf = load_fragB(&Vt[dt * 16 * 72 + ks * 32], 0, 72);
          oacc[dt] = wmma_bf16(pf, vf, oacc[dt]);
        }
      }
    }
    __syncthreads();
  }

  if (lane < 16) sumS[wave * 16 + lane] = lrow;
  __syncthreads();
  int b = bh / H_DIM, h = bh % H_DIM;
  int r0 = (lane >> 4) << 3;
  int coln = lane & 15;
#pragma unroll
  for (int dt = 0; dt < 2; ++dt) {
#pragma unroll
    for (int r = 0; r < 8; ++r) {
      float inv = 1.f / sumS[wave * 16 + r0 + r];
      int n = qRow0 + r0 + r;
      size_t off = ((size_t)b * N_DIM + n) * D_DIM + h * 32 + dt * 16 + coln;
      Out[off] = (bf16)(oacc[dt][r] * inv);
    }
  }
}

// ---------------------------------------------------------------- orchestration
extern "C" void kernel_launch(void* const* d_in, const int* in_sizes, int n_in,
                              void* d_out, int out_size, void* d_ws, size_t ws_size,
                              hipStream_t stream) {
  const float* x_in   = (const float*)d_in[0];
  const int*   eidx   = (const int*)d_in[1];
  const float* ew     = (const float*)d_in[2];
  const float* edge_w = (const float*)d_in[5];
  const float* edge_b = (const float*)d_in[6];
  const float* upd_w1 = (const float*)d_in[7];
  const float* upd_b1 = (const float*)d_in[8];
  const float* upd_w2 = (const float*)d_in[9];
  const float* upd_b2 = (const float*)d_in[10];
  const float* inp_w  = (const float*)d_in[11];
  const float* inp_b  = (const float*)d_in[12];
  const float* out_w  = (const float*)d_in[13];
  const float* out_b  = (const float*)d_in[14];
  const float* ffn_w1 = (const float*)d_in[15];
  const float* ffn_b1 = (const float*)d_in[16];
  const float* ffn_w2 = (const float*)d_in[17];
  const float* ffn_b2 = (const float*)d_in[18];
  const float* ln1_g  = (const float*)d_in[19];
  const float* ln1_b  = (const float*)d_in[20];
  const float* ln2_g  = (const float*)d_in[21];
  const float* ln2_b  = (const float*)d_in[22];
  const float* ln3_g  = (const float*)d_in[23];
  const float* ln3_b  = (const float*)d_in[24];

  const size_t NND = (size_t)NN_DIM * D_DIM;
  char* base = (char*)d_ws;
  size_t off = 0;
  auto alloc = [&](size_t bytes) -> void* {
    void* p = base + off;
    off = off + ((bytes + 255) & ~(size_t)255);
    return p;
  };
  float* qkv_f = (float*)alloc(NND * 3 * sizeof(float));  // qkv / o / f2
  float* agg_f = (float*)alloc(NND * sizeof(float));      // aggX / h2
  float* x1_f  = (float*)alloc(NND * sizeof(float));
  float* x2_f  = (float*)alloc(NND * sizeof(float));
  float* sumew = (float*)alloc(NN_DIM * sizeof(float));
  float* cnt   = (float*)alloc(NN_DIM * sizeof(float));
  bf16* R1 = (bf16*)alloc(NND * sizeof(bf16));            // h0 / xa / x2b
  bf16* R2 = (bf16*)alloc(NND * sizeof(bf16));            // h1 / attn
  bf16* R3 = (bf16*)alloc(NND * 3 * sizeof(bf16));        // q,k,v / f1
  bf16* wU1 = (bf16*)alloc((size_t)D_DIM * D_DIM * sizeof(bf16));
  bf16* wU2 = (bf16*)alloc((size_t)D_DIM * D_DIM * sizeof(bf16));
  bf16* wIP = (bf16*)alloc((size_t)3 * D_DIM * D_DIM * sizeof(bf16));
  bf16* wOP = (bf16*)alloc((size_t)D_DIM * D_DIM * sizeof(bf16));
  bf16* wF1 = (bf16*)alloc((size_t)2 * D_DIM * D_DIM * sizeof(bf16));
  bf16* wF2 = (bf16*)alloc((size_t)2 * D_DIM * D_DIM * sizeof(bf16));

  // 0) zero accumulators
  zero_f<<<(unsigned)((NND + 255) / 256), 256, 0, stream>>>(agg_f, NND);
  zero_f<<<(NN_DIM + 255) / 256, 256, 0, stream>>>(sumew, NN_DIM);
  zero_f<<<(NN_DIM + 255) / 256, 256, 0, stream>>>(cnt, NN_DIM);

  // 1) weight conversion to bf16
  f2b_kernel<<<256, 256, 0, stream>>>(upd_w1, wU1, D_DIM * D_DIM);
  f2b_kernel<<<256, 256, 0, stream>>>(upd_w2, wU2, D_DIM * D_DIM);
  f2b_kernel<<<768, 256, 0, stream>>>(inp_w, wIP, 3 * D_DIM * D_DIM);
  f2b_kernel<<<256, 256, 0, stream>>>(out_w, wOP, D_DIM * D_DIM);
  f2b_kernel<<<512, 256, 0, stream>>>(ffn_w1, wF1, 2 * D_DIM * D_DIM);
  f2b_kernel<<<512, 256, 0, stream>>>(ffn_w2, wF2, 2 * D_DIM * D_DIM);

  // 2) GINE aggregation
  edge_scalar_kernel<<<E_DIM / 256, 256, 0, stream>>>(eidx, ew, sumew, cnt);
  edge_scatter_kernel<<<E_DIM, 256, 0, stream>>>(eidx, x_in, agg_f);
  make_h0_kernel<<<NN_DIM, 256, 0, stream>>>(agg_f, sumew, cnt, edge_w, edge_b,
                                             x_in, R1);

  // 3) node-update MLP: h1 = relu(h0 W1^T + b1); h2 = h1 W2^T + b2
  gemm_bf16_kernel<<<dim3(D_DIM / 128, NN_DIM / 128), 256, GEMM_LDS_BYTES, stream>>>(
      R1, wU1, upd_b1, nullptr, R2, NN_DIM, D_DIM, D_DIM, 1);
  gemm_bf16_kernel<<<dim3(D_DIM / 128, NN_DIM / 128), 256, GEMM_LDS_BYTES, stream>>>(
      R2, wU2, upd_b2, agg_f, nullptr, NN_DIM, D_DIM, D_DIM, 0);

  // 4) x1 = LN(x + h2)
  residual_ln_kernel<<<NN_DIM, 256, 0, stream>>>(x_in, agg_f, ln1_g, ln1_b, x1_f,
                                                 nullptr);

  // 5) PE add, QKV projection, head repack (q pre-scaled by 1/sqrt(dh))
  pe_add_kernel<<<NN_DIM, 256, 0, stream>>>(x1_f, R1);
  gemm_bf16_kernel<<<dim3(3 * D_DIM / 128, NN_DIM / 128), 256, GEMM_LDS_BYTES,
                     stream>>>(R1, wIP, inp_b, qkv_f, nullptr, NN_DIM, 3 * D_DIM,
                               D_DIM, 0);
  repack_qkv_kernel<<<NN_DIM, 256, 0, stream>>>(qkv_f, R3, R3 + NND, R3 + 2 * NND);

  // 6) flash attention -> bf16 attn (B,N,H*dh)
  attn_kernel<<<dim3(B_DIM * H_DIM, N_DIM / 128), 256, ATTN_LDS_BYTES, stream>>>(
      R3, R3 + NND, R3 + 2 * NND, R2);

  // 7) out-proj, x2 = LN(x1 + o)   (o reuses qkv_f)
  gemm_bf16_kernel<<<dim3(D_DIM / 128, NN_DIM / 128), 256, GEMM_LDS_BYTES, stream>>>(
      R2, wOP, out_b, qkv_f, nullptr, NN_DIM, D_DIM, D_DIM, 0);
  residual_ln_kernel<<<NN_DIM, 256, 0, stream>>>(x1_f, qkv_f, ln2_g, ln2_b, x2_f,
                                                 R1);

  // 8) FFN: f1 = gelu(x2 W1^T + b1) (bf16, reuses R3); f2 = f1 W2^T + b2
  gemm_bf16_kernel<<<dim3(2 * D_DIM / 128, NN_DIM / 128), 256, GEMM_LDS_BYTES,
                     stream>>>(R1, wF1, ffn_b1, nullptr, R3, NN_DIM, 2 * D_DIM,
                               D_DIM, 2);
  gemm_bf16_kernel<<<dim3(D_DIM / 128, NN_DIM / 128), 256, GEMM_LDS_BYTES, stream>>>(
      R3, wF2, ffn_b2, qkv_f, nullptr, NN_DIM, D_DIM, 2 * D_DIM, 0);

  // 9) out = LN(x2 + f2)
  residual_ln_kernel<<<NN_DIM, 256, 0, stream>>>(x2_f, qkv_f, ln3_g, ln3_b,
                                                 (float*)d_out, nullptr);

  (void)in_sizes; (void)n_in; (void)out_size; (void)ws_size;
}
